// SimpleVDforPreGate_48524540510487
// MI455X (gfx1250) — compile-verified, matrix operand order checked by reading, not run on hardware
//
#include <hip/hip_runtime.h>

#define BB   16
#define LL   576
#define CIN  1024
#define DD   768
#define KK   8192
#define NROW (BB*LL)   // 9216

typedef __attribute__((ext_vector_type(16))) __bf16       v16bf;
typedef __attribute__((ext_vector_type(8)))  float        v8f;
typedef __attribute__((ext_vector_type(4)))  unsigned int v4u;

union Frag { v16bf v; v4u q[2]; };

__device__ __forceinline__ unsigned short f2bf(float f) {
    union { float f; unsigned u; } x; x.f = f;
    unsigned u = x.u;
    unsigned rb = (u >> 16) & 1u;
    u += 0x7fffu + rb;                 // round-to-nearest-even
    return (unsigned short)(u >> 16);
}

// ---------------------------------------------------------------- fp32 -> bf16
__global__ void k_cvt_bf16(const float* __restrict__ src,
                           unsigned short* __restrict__ dst, int n) {
    int i = blockIdx.x * blockDim.x + threadIdx.x;
    if (i < n) dst[i] = f2bf(src[i]);
}

// ---------------------------------------------------------------- block reduce
__device__ __forceinline__ float block_sum(float v, float* sm) {
    for (int o = 16; o; o >>= 1) v += __shfl_down(v, o, 32);
    int lane = threadIdx.x & 31, w = threadIdx.x >> 5;
    if (lane == 0) sm[w] = v;
    __syncthreads();
    float r = (threadIdx.x < 8) ? sm[threadIdx.x] : 0.f;
    if (threadIdx.x < 8)
        for (int o = 4; o; o >>= 1) r += __shfl_down(r, o, 32);
    if (threadIdx.x == 0) sm[0] = r;
    __syncthreads();
    float out = sm[0];
    __syncthreads();
    return out;
}

// ---------------------------------------------------------------- row |e|^2
__global__ void k_rownorm2(const float* __restrict__ E, float* __restrict__ en) {
    __shared__ float sm[8];
    long row = blockIdx.x;
    float s = 0.f;
    for (int j = 0; j < 3; ++j) {
        float x = E[row * DD + threadIdx.x + j * 256];
        s += x * x;
    }
    s = block_sum(s, sm);
    if (threadIdx.x == 0) en[row] = s;
}

// ---------------------------------------------------------------- WMMA GEMM
// C[m][n] = sum_k A[m][k]*B[n][k] + bias[n]   (bf16 in, f32 out)
// block = 256 (8 waves, 2x4). WG tile = 64(M) x 64(N); each wave 32x16 with
// two accumulators sharing every B fragment. Optional A-row gather.
__global__ void k_gemm_bias(const unsigned short* __restrict__ A,
                            const unsigned short* __restrict__ B,
                            const float* __restrict__ bias,
                            float* __restrict__ C,
                            int N, int K, const int* __restrict__ aidx) {
    int lane = threadIdx.x & 31, wave = threadIdx.x >> 5;
    int wm = wave & 1, wn = wave >> 1;
    int m0 = blockIdx.y * 64 + wm * 32;
    int n0 = blockIdx.x * 64 + wn * 16;
    int r = lane & 15, g = lane >> 4;

    long ar0 = m0 + r, ar1 = m0 + 16 + r;
    if (aidx) { ar0 = aidx[ar0]; ar1 = aidx[ar1]; }
    const unsigned short* ap0 = A + ar0 * (long)K + g * 8;
    const unsigned short* ap1 = A + ar1 * (long)K + g * 8;
    const unsigned short* bp  = B + (long)(n0 + r) * K + g * 8;

    v8f acc0 = {}, acc1 = {};
    for (int k = 0; k < K; k += 32) {
        Frag a0, a1, b;
        a0.q[0] = *(const v4u*)(ap0 + k);
        a0.q[1] = *(const v4u*)(ap0 + k + 16);
        a1.q[0] = *(const v4u*)(ap1 + k);
        a1.q[1] = *(const v4u*)(ap1 + k + 16);
        b.q[0]  = *(const v4u*)(bp + k);
        b.q[1]  = *(const v4u*)(bp + k + 16);
        __builtin_prefetch((const void*)(bp + k + 128), 0, 1);
        acc0 = __builtin_amdgcn_wmma_f32_16x16x32_bf16(
                   false, a0.v, false, b.v, (short)0, acc0, false, false);
        acc1 = __builtin_amdgcn_wmma_f32_16x16x32_bf16(
                   false, a1.v, false, b.v, (short)0, acc1, false, false);
    }
    float bz = bias ? bias[n0 + r] : 0.f;
#pragma unroll
    for (int t = 0; t < 8; ++t) {
        int row0 = m0 + t + g * 8;
        C[(long)row0 * N + n0 + r]        = acc0[t] + bz;
        C[(long)(row0 + 16) * N + n0 + r] = acc1[t] + bz;
    }
}

// ---------------------------------------------------------------- LN + ReLU
__global__ void k_ln_relu(const float* __restrict__ X,
                          const float* __restrict__ w, const float* __restrict__ b,
                          float* __restrict__ Yf, unsigned short* __restrict__ Yb) {
    __shared__ float sm[8];
    long row = blockIdx.x;
    float v[3], s = 0.f, s2 = 0.f;
#pragma unroll
    for (int j = 0; j < 3; ++j) {
        float x = X[row * DD + threadIdx.x + j * 256];
        v[j] = x; s += x; s2 += x * x;
    }
    s  = block_sum(s, sm);
    s2 = block_sum(s2, sm);
    float mu = s / DD, var = s2 / DD - mu * mu;
    float inv = rsqrtf(var + 1e-5f);
#pragma unroll
    for (int j = 0; j < 3; ++j) {
        int c = threadIdx.x + j * 256;
        float y = (v[j] - mu) * inv * w[c] + b[c];
        y = fmaxf(y, 0.f);
        Yf[row * DD + c] = y;
        Yb[row * DD + c] = f2bf(y);
    }
}

// ---------------------------------------------------------------- fused VQ
// score[m][n] = |e_n|^2 - 2 * x_m . e_n ; per-row top-10 smallest + argmin.
// block = 256 (8 waves). 64 rows per block (4 accumulators per wave share
// each B fragment -> half the L2 traffic on embed). A tile staged to LDS
// with async global->LDS DMA (ASYNCcnt path).
#define APAD 776   // 768 + 8 ushorts padding (bank-conflict-free frag reads)
#define VQR  64
__global__ void k_vq(const unsigned short* __restrict__ Xb,
                     const unsigned short* __restrict__ Eb,
                     const float* __restrict__ en,
                     int* __restrict__ idx_out,
                     float* __restrict__ topk_logit,
                     int* __restrict__ topk_idx) {
    __shared__ __align__(16) unsigned short As[VQR * APAD];   // 99,328 B
    __shared__ float stile[8][VQR][16];                       // 32,768 B

    int lane = threadIdx.x & 31, w = threadIdx.x >> 5;
    int r = lane & 15, g = lane >> 4;
    long mbase = (long)blockIdx.x * VQR;

    // stage 64x768 A rows into LDS via async global->LDS b128 DMA
    for (int it = threadIdx.x; it < VQR * 96; it += 256) {
        int rr = it / 96, ch = it % 96;
        unsigned lds_off = (unsigned)(size_t)(&As[rr * APAD + ch * 8]);
        const unsigned short* gp = Xb + (mbase + rr) * DD + ch * 8;
        asm volatile("global_load_async_to_lds_b128 %0, %1, off"
                     :: "v"(lds_off), "v"(gp) : "memory");
    }
    asm volatile("s_wait_asynccnt 0" ::: "memory");
    __syncthreads();

    float tv0[10], tv1[10]; int ti0[10], ti1[10];
#pragma unroll
    for (int j = 0; j < 10; ++j) {
        tv0[j] = 3.4e38f; ti0[j] = 0; tv1[j] = 3.4e38f; ti1[j] = 0;
    }

    for (int tt = 0; tt < 64; ++tt) {
        int n0 = (w + 8 * tt) * 16;
        const unsigned short* bp = Eb + (long)(n0 + r) * DD + g * 8;
        v8f c0 = {}, c1 = {}, c2 = {}, c3 = {};
        for (int k = 0; k < DD; k += 32) {
            Frag a0, a1, a2, a3, b;
            a0.q[0] = *(const v4u*)(&As[(r)      * APAD + k + g * 8]);
            a0.q[1] = *(const v4u*)(&As[(r)      * APAD + k + g * 8 + 16]);
            a1.q[0] = *(const v4u*)(&As[(16 + r) * APAD + k + g * 8]);
            a1.q[1] = *(const v4u*)(&As[(16 + r) * APAD + k + g * 8 + 16]);
            a2.q[0] = *(const v4u*)(&As[(32 + r) * APAD + k + g * 8]);
            a2.q[1] = *(const v4u*)(&As[(32 + r) * APAD + k + g * 8 + 16]);
            a3.q[0] = *(const v4u*)(&As[(48 + r) * APAD + k + g * 8]);
            a3.q[1] = *(const v4u*)(&As[(48 + r) * APAD + k + g * 8 + 16]);
            b.q[0]  = *(const v4u*)(bp + k);
            b.q[1]  = *(const v4u*)(bp + k + 16);
            __builtin_prefetch((const void*)(bp + k + 128), 0, 1);
            c0 = __builtin_amdgcn_wmma_f32_16x16x32_bf16(
                     false, a0.v, false, b.v, (short)0, c0, false, false);
            c1 = __builtin_amdgcn_wmma_f32_16x16x32_bf16(
                     false, a1.v, false, b.v, (short)0, c1, false, false);
            c2 = __builtin_amdgcn_wmma_f32_16x16x32_bf16(
                     false, a2.v, false, b.v, (short)0, c2, false, false);
            c3 = __builtin_amdgcn_wmma_f32_16x16x32_bf16(
                     false, a3.v, false, b.v, (short)0, c3, false, false);
        }
        float e_n = en[n0 + r];
#pragma unroll
        for (int t = 0; t < 8; ++t) {
            stile[w][t + g * 8][r]      = e_n - 2.0f * c0[t];
            stile[w][16 + t + g * 8][r] = e_n - 2.0f * c1[t];
            stile[w][32 + t + g * 8][r] = e_n - 2.0f * c2[t];
            stile[w][48 + t + g * 8][r] = e_n - 2.0f * c3[t];
        }
        // wave-local scan: lane owns rows lane and lane+32 of this wave's cols
        for (int c = 0; c < 16; ++c) {
            int n = n0 + c;
            float s0 = stile[w][lane][c];
            if (s0 < tv0[9]) {
                float cv = s0; int ci = n;
#pragma unroll
                for (int p = 0; p < 10; ++p) {
                    if (cv < tv0[p]) {
                        float tf = tv0[p]; tv0[p] = cv; cv = tf;
                        int   tg = ti0[p]; ti0[p] = ci; ci = tg;
                    }
                }
            }
            float s1 = stile[w][32 + lane][c];
            if (s1 < tv1[9]) {
                float cv = s1; int ci = n;
#pragma unroll
                for (int p = 0; p < 10; ++p) {
                    if (cv < tv1[p]) {
                        float tf = tv1[p]; tv1[p] = cv; cv = tf;
                        int   tg = ti1[p]; ti1[p] = ci; ci = tg;
                    }
                }
            }
        }
    }

    // overlay per-wave top-10 lists on the (now dead) A-stage LDS
    __syncthreads();
    float* mvalp = (float*)As;                         // [8][64][10] = 20480 B
    int*   midxp = (int*)((char*)As + 8 * VQR * 10 * 4);
#pragma unroll
    for (int j = 0; j < 10; ++j) {
        mvalp[(w * VQR + lane) * 10 + j]      = tv0[j];
        midxp[(w * VQR + lane) * 10 + j]      = ti0[j];
        mvalp[(w * VQR + 32 + lane) * 10 + j] = tv1[j];
        midxp[(w * VQR + 32 + lane) * 10 + j] = ti1[j];
    }
    __syncthreads();

    if (w < 2) {              // 64 rows -> waves 0,1; one row per lane
        int row = w * 32 + lane;
        float bv[10]; int bi[10];
#pragma unroll
        for (int j = 0; j < 10; ++j) { bv[j] = 3.4e38f; bi[j] = 0; }
        for (int ww = 0; ww < 8; ++ww)
            for (int j = 0; j < 10; ++j) {
                float cv = mvalp[(ww * VQR + row) * 10 + j];
                int   ci = midxp[(ww * VQR + row) * 10 + j];
                if (cv < bv[9]) {
#pragma unroll
                    for (int p = 0; p < 10; ++p) {
                        if (cv < bv[p]) {
                            float tf = bv[p]; bv[p] = cv; cv = tf;
                            int   tg = bi[p]; bi[p] = ci; ci = tg;
                        }
                    }
                }
            }
        long grow = mbase + row;
        idx_out[grow] = bi[0];
#pragma unroll
        for (int j = 0; j < 10; ++j) {
            topk_logit[grow * 10 + j] = -bv[j];   // -score; row const cancels
            topk_idx[grow * 10 + j]   = bi[j];
        }
    }
}

// ---------------------------------------------------------------- gate scores
__global__ void k_gate(const float* __restrict__ emb_pt, const float* __restrict__ xq,
                       const float* __restrict__ gw, const float* __restrict__ gb,
                       const float* __restrict__ glw, const float* __restrict__ glb,
                       float* __restrict__ scores) {
    int wave = threadIdx.x >> 5, lane = threadIdx.x & 31;
    long row = (long)blockIdx.x * 8 + wave;
    const float* e = emb_pt + row * DD;
    const float* x = xq + row * DD;
    float g0 = 0.f, g1 = 0.f;
    for (int i = lane; i < DD; i += 32) {
        float ev = e[i], xv = x[i];
        g0 += ev * gw[i]        + xv * gw[DD + i];
        g1 += ev * gw[1536 + i] + xv * gw[1536 + DD + i];
    }
    for (int o = 16; o; o >>= 1) { g0 += __shfl_down(g0, o, 32); g1 += __shfl_down(g1, o, 32); }
    if (lane == 0) {
        g0 += gb[0]; g1 += gb[1];
        float mu  = 0.5f * (g0 + g1);
        float var = 0.5f * ((g0 - mu) * (g0 - mu) + (g1 - mu) * (g1 - mu));
        float inv = rsqrtf(var + 1e-5f);
        float y0 = fmaxf(0.f, (g0 - mu) * inv * glw[0] + glb[0]);
        float y1 = fmaxf(0.f, (g1 - mu) * inv * glw[1] + glb[1]);
        float m = fmaxf(y0, y1);
        float e0 = __expf(y0 - m), e1 = __expf(y1 - m);
        float d = e0 + e1;
        scores[row * 2 + 0] = e0 / d;
        scores[row * 2 + 1] = e1 / d;
    }
}

// ---------------------------------------------------------------- masked negatives
__global__ void k_negprep(const int* __restrict__ maskI, const int* __restrict__ idx,
                          const float* __restrict__ tkl, const int* __restrict__ tki,
                          int* __restrict__ masked_label, int* __restrict__ neg_at_mask) {
    int b = threadIdx.x;
    if (b >= BB) return;
    long row = (long)b * LL + maskI[b];
    masked_label[b] = idx[row];
    float mx = -3.4e38f, p[10], s = 0.f;
    for (int j = 0; j < 10; ++j) mx = fmaxf(mx, tkl[row * 10 + j]);
    for (int j = 0; j < 10; ++j) { p[j] = __expf(tkl[row * 10 + j] - mx); s += p[j]; }
    unsigned h = (unsigned)row * 2654435761u ^ 0x9E3779B9u;
    h ^= h >> 16; h *= 0x85ebca6bu; h ^= h >> 13; h *= 0xc2b2ae35u; h ^= h >> 16;
    float u = (h >> 8) * (1.f / 16777216.f) * s;
    float acc = 0.f; int samp = 9;
    for (int j = 0; j < 10; ++j) { acc += p[j]; if (u < acc) { samp = j; break; } }
    neg_at_mask[b] = tki[row * 10 + samp];
}

// ---------------------------------------------------------------- final fuse:
// gating mix + sine pos-enc + two LayerNorms -> out_pos, out_neg
__global__ void k_final(const float* __restrict__ emb_pt, const float* __restrict__ xq,
                        const float* __restrict__ embed, const int* __restrict__ idx,
                        const int* __restrict__ masked_label,
                        const int* __restrict__ neg_at_mask,
                        const float* __restrict__ scores,
                        const float* __restrict__ ow, const float* __restrict__ ob,
                        float* __restrict__ out) {
    __shared__ float sm[8];
    long row = blockIdx.x;
    int b = (int)(row / LL), l = (int)(row % LL);
    float es = scores[row * 2 + 0], is = scores[row * 2 + 1];
    int myidx = idx[row];
    long nidx = (myidx == masked_label[b]) ? neg_at_mask[b] : myidx;
    int yi = l / 24, xi = l % 24;
    const float tp = 6.283185307179586f / (24.0f + 1e-6f);
    float ynorm = (yi + 1) * tp, xnorm = (xi + 1) * tp;
    float av[3], bv[3], sa = 0, sa2 = 0, sb = 0, sb2 = 0;
#pragma unroll
    for (int j = 0; j < 3; ++j) {
        int c = threadIdx.x + j * 256;
        int p = (c < 384) ? (c >> 1) : ((c - 384) >> 1);
        float base = (c < 384) ? ynorm : xnorm;
        float arg = base * __expf(-(float)p * 0.04797052313f);  // 1/10000^(p/192)
        float pos = (c & 1) ? __cosf(arg) : __sinf(arg);
        float xv = xq[row * DD + c];
        float a = emb_pt[row * DD + c] * es + xv * is + pos;
        float n = embed[nidx * DD + c] * es + xv * is + pos;
        av[j] = a; bv[j] = n;
        sa += a; sa2 += a * a; sb += n; sb2 += n * n;
    }
    sa = block_sum(sa, sm); sa2 = block_sum(sa2, sm);
    sb = block_sum(sb, sm); sb2 = block_sum(sb2, sm);
    float mua = sa / DD, ia = rsqrtf(sa2 / DD - mua * mua + 1e-5f);
    float mub = sb / DD, ib = rsqrtf(sb2 / DD - mub * mub + 1e-5f);
#pragma unroll
    for (int j = 0; j < 3; ++j) {
        int c = threadIdx.x + j * 256;
        out[row * DD + c]                   = (av[j] - mua) * ia * ow[c] + ob[c];
        out[(long)NROW * DD + row * DD + c] = (bv[j] - mub) * ib * ow[c] + ob[c];
    }
}

// ---------------------------------------------------------------- launch
extern "C" void kernel_launch(void* const* d_in, const int* in_sizes, int n_in,
                              void* d_out, int out_size, void* d_ws, size_t ws_size,
                              hipStream_t stream) {
    const float* img   = (const float*)d_in[0];
    const int*   maskI = (const int*)  d_in[1];
    const float* lin_w = (const float*)d_in[2];
    const float* lin_b = (const float*)d_in[3];
    const float* ln1w  = (const float*)d_in[4];
    const float* ln1b  = (const float*)d_in[5];
    const float* embed = (const float*)d_in[6];
    const float* posw  = (const float*)d_in[7];
    const float* posb  = (const float*)d_in[8];
    const float* gw    = (const float*)d_in[9];
    const float* gb    = (const float*)d_in[10];
    const float* glw   = (const float*)d_in[11];
    const float* glb   = (const float*)d_in[12];
    const float* ow    = (const float*)d_in[13];
    const float* ob    = (const float*)d_in[14];
    float* out = (float*)d_out;

    char* ws = (char*)d_ws;
    size_t o = 0;
    auto alloc = [&](size_t bytes) -> void* {
        void* p = ws + o; o += (bytes + 255) & ~(size_t)255; return p;
    };
    unsigned short* img_b   = (unsigned short*)alloc((size_t)NROW * CIN * 2);
    unsigned short* linw_b  = (unsigned short*)alloc((size_t)DD * CIN * 2);
    unsigned short* embed_b = (unsigned short*)alloc((size_t)KK * DD * 2);
    unsigned short* posw_b  = (unsigned short*)alloc((size_t)DD * DD * 2);
    float*          lin_out = (float*)alloc((size_t)NROW * DD * 4);
    float*          xq_f    = (float*)alloc((size_t)NROW * DD * 4);
    unsigned short* xq_b    = (unsigned short*)alloc((size_t)NROW * DD * 2);
    float*          emb_pt  = (float*)alloc((size_t)NROW * DD * 4);
    float*          enorm   = (float*)alloc((size_t)KK * 4);
    int*            idx     = (int*)alloc((size_t)NROW * 4);
    float*          tkl     = (float*)alloc((size_t)NROW * 10 * 4);
    int*            tki     = (int*)alloc((size_t)NROW * 10 * 4);
    float*          scores  = (float*)alloc((size_t)NROW * 2 * 4);
    int*            mlab    = (int*)alloc(64);
    int*            negm    = (int*)alloc(64);

    // fp32 -> bf16 conversions
    {
        int n0 = NROW * CIN, n1 = DD * CIN, n2 = KK * DD, n3 = DD * DD;
        k_cvt_bf16<<<(n0 + 255) / 256, 256, 0, stream>>>(img,   img_b,   n0);
        k_cvt_bf16<<<(n1 + 255) / 256, 256, 0, stream>>>(lin_w, linw_b,  n1);
        k_cvt_bf16<<<(n2 + 255) / 256, 256, 0, stream>>>(embed, embed_b, n2);
        k_cvt_bf16<<<(n3 + 255) / 256, 256, 0, stream>>>(posw,  posw_b,  n3);
    }
    k_rownorm2<<<KK, 256, 0, stream>>>(embed, enorm);

    // GEMM1: img @ lin_w.T + lin_b
    k_gemm_bias<<<dim3(DD / 64, NROW / 64), 256, 0, stream>>>(
        img_b, linw_b, lin_b, lin_out, DD, CIN, nullptr);
    // LN1 + ReLU -> xq (f32 + bf16)
    k_ln_relu<<<NROW, 256, 0, stream>>>(lin_out, ln1w, ln1b, xq_f, xq_b);
    // fused VQ: distances + argmin + top-10
    k_vq<<<NROW / VQR, 256, 0, stream>>>(xq_b, embed_b, enorm, idx, tkl, tki);
    // GEMM3: embed[idx] @ pos_w.T + pos_b  (row gather)
    k_gemm_bias<<<dim3(DD / 64, NROW / 64), 256, 0, stream>>>(
        embed_b, posw_b, posb, emb_pt, DD, DD, idx);
    // gate softmax scores
    k_gate<<<NROW / 8, 256, 0, stream>>>(emb_pt, xq_f, gw, gb, glw, glb, scores);
    // masked negative selection + categorical sample (16 rows)
    k_negprep<<<1, 32, 0, stream>>>(maskI, idx, tkl, tki, mlab, negm);
    // final mix + pos-enc + dual LayerNorm
    k_final<<<NROW, 256, 0, stream>>>(emb_pt, xq_f, embed, idx, mlab, negm,
                                      scores, ow, ob, out);
    (void)in_sizes; (void)n_in; (void)out_size; (void)ws_size;
}
